// GraphAttentionNetwork_13855564497185
// MI455X (gfx1250) — compile-verified, matrix operand order checked by reading, not run on hardware
//
#include <hip/hip_runtime.h>
#include <hip/hip_bf16.h>

typedef __attribute__((ext_vector_type(16))) __bf16 v16bf;
typedef __attribute__((ext_vector_type(8)))  __bf16 v8bf;
typedef __attribute__((ext_vector_type(8)))  float  v8f;

#define NEG_SLOPE 0.2f

static __host__ __device__ inline int cdiv(int a, int b) { return (a + b - 1) / b; }

// ---------- monotone float<->uint for atomic segment-max ----------
__device__ __forceinline__ unsigned f2ord(float f) {
  unsigned u = __float_as_uint(f);
  return (u & 0x80000000u) ? ~u : (u | 0x80000000u);
}
__device__ __forceinline__ float ord2f(unsigned k) {
  unsigned u = (k & 0x80000000u) ? (k & 0x7FFFFFFFu) : ~k;
  return __uint_as_float(u);
}

// -------- fp32 -> bf16 conversion pass --------
__global__ void k_f32_to_bf16(const float* __restrict__ in, __bf16* __restrict__ o, int n) {
  int i = blockIdx.x * blockDim.x + threadIdx.x;
  if (i < n) o[i] = (__bf16)in[i];
}

// -------- pack fp32 weight [K,N] into WMMA B-fragment order --------
// layout: Bp[((kt*ntiles + nt)*32 + lane)*16 + i]  (16 contiguous bf16 per lane)
__global__ void k_pack_b(const float* __restrict__ W, __bf16* __restrict__ Bp, int K, int N) {
  int t = blockIdx.x * blockDim.x + threadIdx.x;
  int ntiles = N / 16, ktiles = K / 32;
  if (t >= ktiles * ntiles * 32) return;
  int lane = t & 31, tile = t >> 5;
  int nt = tile % ntiles, kt = tile / ntiles;
  int col = nt * 16 + (lane & 15);
  int kb = (lane < 16) ? 0 : 8;
  __bf16* o = Bp + (size_t)t * 16;
#pragma unroll
  for (int i = 0; i < 8; ++i) {
    o[i]     = (__bf16)W[(size_t)(kt * 32 + kb + i) * N + col];
    o[8 + i] = (__bf16)W[(size_t)(kt * 32 + kb + 16 + i) * N + col];
  }
}

// ---------------------------------------------------------------
// WMMA GEMM on pre-converted bf16:  C[M,N] = act(A @ Bp + bias)
// A: row-major bf16 [M,K]; Bp: fragment-packed bf16.
// 256-thread blocks = 8 waves; wave w -> row tile blockIdx.x*8+w.
// ---------------------------------------------------------------
__global__ void k_gemm_bf16(const __bf16* __restrict__ A, const __bf16* __restrict__ Bp,
                            const float* __restrict__ bias, float* __restrict__ C,
                            int Mtiles, int K, int N, int mode) {
  const int lane = threadIdx.x & 31;
  const int rt   = blockIdx.x * 8 + (threadIdx.x >> 5);
  if (rt >= Mtiles) return;                 // wave-uniform; EXEC stays all-1
  const int nt     = blockIdx.y;
  const int ntiles = N / 16;
  const int kb   = (lane < 16) ? 0 : 8;
  const int roff = (lane < 16) ? 0 : 8;
  const int gRow = rt * 16 + (lane & 15);
  const int gCol = nt * 16 + (lane & 15);

  const __bf16* ap = A + (size_t)gRow * K;
  v8f acc = {};
  for (int k0 = 0; k0 < K; k0 += 32) {
    if (k0 + 32 < K) __builtin_prefetch(ap + k0 + 32 + kb, 0, 1);   // global_prefetch_b8
    union { v16bf v; v8bf h[2]; } a;
    a.h[0] = *(const v8bf*)(ap + k0 + kb);        // 16B aligned
    a.h[1] = *(const v8bf*)(ap + k0 + kb + 16);
    const v16bf b = *(const v16bf*)(Bp + ((size_t)((k0 >> 5) * ntiles + nt) * 32 + lane) * 16);
    acc = __builtin_amdgcn_wmma_f32_16x16x32_bf16(false, a.v, false, b,
                                                  (short)0, acc, false, false);
  }
  const float bv = bias ? bias[gCol] : 0.f;
#pragma unroll
  for (int r = 0; r < 8; ++r) {
    float v = acc[r] + bv;
    if (mode == 1) v = v > 0.f ? v : 0.f;
    C[(size_t)(rt * 16 + r + roff) * N + gCol] = v;
  }
}

// -------- attention coefficients: el/er[n,h] = sum_d feat*al/ar --------
__global__ void k_attn(const float* __restrict__ feat, const float* __restrict__ al,
                       const float* __restrict__ ar, float* __restrict__ el,
                       float* __restrict__ er, int n, int heads, int d) {
  int idx = blockIdx.x * blockDim.x + threadIdx.x;
  if (idx >= n * heads) return;
  int node = idx / heads, h = idx % heads;
  const float* f = feat + (size_t)node * heads * d + h * d;
  float sl = 0.f, sr = 0.f;
  for (int i = 0; i < d; ++i) { sl += f[i] * al[h * d + i]; sr += f[i] * ar[h * d + i]; }
  el[idx] = sl; er[idx] = sr;
}

__global__ void k_zero(float* __restrict__ p, int n) {
  int i = blockIdx.x * blockDim.x + threadIdx.x;
  if (i < n) p[i] = 0.f;
}

// -------- edge pass 1: e = leaky_relu(el[src]+er[dst]); segment max --------
__global__ void k_edge_max(const float* __restrict__ el, const float* __restrict__ er,
                           const int* __restrict__ src, const int* __restrict__ dst,
                           float* __restrict__ ebuf, unsigned* __restrict__ menc,
                           int E, int heads) {
  int idx = blockIdx.x * blockDim.x + threadIdx.x;
  if (idx >= E * heads) return;
  int e = idx / heads, h = idx % heads;
  float v = el[src[e] * heads + h] + er[dst[e] * heads + h];
  v = v > 0.f ? v : NEG_SLOPE * v;
  ebuf[idx] = v;
  atomicMax(&menc[dst[e] * heads + h], f2ord(v));
}

// -------- edge pass 2: ex = exp(e - m[dst]); segment sum --------
__global__ void k_edge_expsum(const float* __restrict__ ebuf, const unsigned* __restrict__ menc,
                              const int* __restrict__ dst, float* __restrict__ exbuf,
                              float* __restrict__ s, int E, int heads) {
  int idx = blockIdx.x * blockDim.x + threadIdx.x;
  if (idx >= E * heads) return;
  int e = idx / heads, h = idx % heads;
  float m = ord2f(menc[dst[e] * heads + h]);
  float ex = __expf(ebuf[idx] - m);
  exbuf[idx] = ex;
  atomicAdd(&s[dst[e] * heads + h], ex);
}

// -------- edge pass 3: rst[dst] += feat[src] * (ex / s[dst]) --------
__global__ void k_edge_aggr(const float* __restrict__ exbuf, const float* __restrict__ s,
                            const int* __restrict__ src, const int* __restrict__ dst,
                            const float* __restrict__ feat, float* __restrict__ rst,
                            int E, int heads, int d) {
  int idx = blockIdx.x * blockDim.x + threadIdx.x;
  if (idx >= E * heads) return;
  int e = idx / heads, h = idx % heads;
  float a = exbuf[idx] / s[dst[e] * heads + h];
  const float* f = feat + (size_t)src[e] * heads * d + h * d;
  float* r = rst + (size_t)dst[e] * heads * d + h * d;
  for (int i = 0; i < d; ++i) atomicAdd(&r[i], f[i] * a);
}

// -------- finalize: h = BN( act(rst + res + b) ) --------
__global__ void k_finalize(const float* __restrict__ rst, const float* __restrict__ res,
                           const float* __restrict__ b, const float* __restrict__ g,
                           const float* __restrict__ bt, const float* __restrict__ mean,
                           const float* __restrict__ var, float* __restrict__ out,
                           int n, int C, int act) {
  int idx = blockIdx.x * blockDim.x + threadIdx.x;
  if (idx >= n * C) return;
  int c = idx % C;
  float x = rst[idx] + res[idx] + b[c];
  if (act) x = x > 0.f ? x : (__expf(x) - 1.f);
  out[idx] = (x - mean[c]) * g[c] * rsqrtf(var[c] + 1e-5f) + bt[c];
}

// ---------------------------------------------------------------
// Fused edge classifier, one wave / 16 edges, bf16 LDS tiles.
//   rep = [h[src](32) | h[dst](32) | relu(edge_bf @ We + be)(128)]
//   z1  = relu(rep @ Wc1 + bc1)  (192->32, WMMA, packed B)
//   z2  = relu(z1 @ Wc2 + bc2)   (32->16, WMMA, packed B)
//   out = z2 @ Wc3 + bc3         (16->2, VALU)
// ---------------------------------------------------------------
__global__ void k_classifier(const float* __restrict__ h, const __bf16* __restrict__ edge_bf,
                             const int* __restrict__ src, const int* __restrict__ dst,
                             const __bf16* __restrict__ Wep, const float* __restrict__ be,
                             const __bf16* __restrict__ Wc1p, const float* __restrict__ bc1,
                             const __bf16* __restrict__ Wc2p, const float* __restrict__ bc2,
                             const float* __restrict__ Wc3, const float* __restrict__ bc3,
                             float* __restrict__ out, int E) {
  __shared__ __align__(16) __bf16 rep[16][192];
  __shared__ __align__(16) __bf16 z1s[16][32];
  __shared__ __align__(16) __bf16 z2s[16][16];

  const int lane = threadIdx.x;
  const int e0   = blockIdx.x * 16;
  const int row  = lane & 15;
  const int kb   = (lane < 16) ? 0 : 8;
  const int roff = (lane < 16) ? 0 : 8;

  // gather h[src] -> rep[:,0:32], h[dst] -> rep[:,32:64]
  for (int t = lane; t < 16 * 64; t += 32) {
    int el_ = t >> 6, c = t & 63;
    int node = (c < 32) ? src[e0 + el_] : dst[e0 + el_];
    rep[el_][c] = (__bf16)h[(size_t)node * 32 + (c & 31)];
  }

  // edge-projection A fragments (16 edges x 64 feats), hoisted over col tiles
  v16bf aef[2];
  {
    const __bf16* ap = edge_bf + (size_t)(e0 + row) * 64;
#pragma unroll
    for (int ks = 0; ks < 2; ++ks) {
      union { v16bf v; v8bf h2[2]; } a;
      a.h2[0] = *(const v8bf*)(ap + ks * 32 + kb);
      a.h2[1] = *(const v8bf*)(ap + ks * 32 + kb + 16);
      aef[ks] = a.v;
    }
  }
  // ef = relu(edge_bf @ We + be) -> rep[:,64:192]   (ntiles=8, ktiles=2)
  for (int ct = 0; ct < 8; ++ct) {
    v8f acc = {};
    const int col = ct * 16 + (lane & 15);
#pragma unroll
    for (int ks = 0; ks < 2; ++ks) {
      const v16bf b = *(const v16bf*)(Wep + ((size_t)(ks * 8 + ct) * 32 + lane) * 16);
      acc = __builtin_amdgcn_wmma_f32_16x16x32_bf16(false, aef[ks], false, b,
                                                    (short)0, acc, false, false);
    }
    const float bv = be[col];
#pragma unroll
    for (int r = 0; r < 8; ++r) {
      float v = acc[r] + bv;
      rep[r + roff][64 + col] = (__bf16)(v > 0.f ? v : 0.f);
    }
  }
  __syncthreads();

  // z1 = relu(rep @ Wc1 + bc1): ktiles=6, ntiles=2
  for (int ct = 0; ct < 2; ++ct) {
    v8f acc = {};
    const int col = ct * 16 + (lane & 15);
    for (int ks = 0; ks < 6; ++ks) {
      union { v16bf v; v8bf h2[2]; } a;
      const int kbase = ks * 32 + kb;
      a.h2[0] = *(const v8bf*)&rep[row][kbase];
      a.h2[1] = *(const v8bf*)&rep[row][kbase + 16];
      const v16bf b = *(const v16bf*)(Wc1p + ((size_t)(ks * 2 + ct) * 32 + lane) * 16);
      acc = __builtin_amdgcn_wmma_f32_16x16x32_bf16(false, a.v, false, b,
                                                    (short)0, acc, false, false);
    }
    const float bv = bc1[col];
#pragma unroll
    for (int r = 0; r < 8; ++r) {
      float v = acc[r] + bv;
      z1s[r + roff][col] = (__bf16)(v > 0.f ? v : 0.f);
    }
  }
  __syncthreads();

  // z2 = relu(z1 @ Wc2 + bc2): ktiles=1, ntiles=1
  {
    v8f acc = {};
    const int col = lane & 15;
    union { v16bf v; v8bf h2[2]; } a;
    a.h2[0] = *(const v8bf*)&z1s[row][kb];
    a.h2[1] = *(const v8bf*)&z1s[row][kb + 16];
    const v16bf b = *(const v16bf*)(Wc2p + (size_t)lane * 16);
    acc = __builtin_amdgcn_wmma_f32_16x16x32_bf16(false, a.v, false, b,
                                                  (short)0, acc, false, false);
    const float bv = bc2[col];
#pragma unroll
    for (int r = 0; r < 8; ++r) {
      float v = acc[r] + bv;
      z2s[r + roff][col] = (__bf16)(v > 0.f ? v : 0.f);
    }
  }
  __syncthreads();

  // out = z2 @ Wc3 + bc3 : 16 edges x 2 outputs == 32 lanes
  {
    const int el_ = lane >> 1, c = lane & 1;
    float s = bc3[c];
#pragma unroll
    for (int k = 0; k < 16; ++k) s += (float)z2s[el_][k] * Wc3[k * 2 + c];
    out[(size_t)(e0 + el_) * 2 + c] = s;
  }
}

// ---------------------------------------------------------------
extern "C" void kernel_launch(void* const* d_in, const int* in_sizes, int n_in,
                              void* d_out, int out_size, void* d_ws, size_t ws_size,
                              hipStream_t stream) {
  const float* node_feats = (const float*)d_in[0];
  const float* edge_feats = (const float*)d_in[1];
  const int*   src        = (const int*)d_in[2];
  const int*   dst        = (const int*)d_in[3];
  const float* W0  = (const float*)d_in[4];
  const float* al0 = (const float*)d_in[5];
  const float* ar0 = (const float*)d_in[6];
  const float* b0  = (const float*)d_in[7];
  const float* g0  = (const float*)d_in[8];
  const float* bt0 = (const float*)d_in[9];
  const float* m0  = (const float*)d_in[10];
  const float* v0  = (const float*)d_in[11];
  const float* W1  = (const float*)d_in[12];
  const float* al1 = (const float*)d_in[13];
  const float* ar1 = (const float*)d_in[14];
  const float* b1  = (const float*)d_in[15];
  const float* Wres1 = (const float*)d_in[16];
  const float* g1  = (const float*)d_in[17];
  const float* bt1 = (const float*)d_in[18];
  const float* m1  = (const float*)d_in[19];
  const float* v1  = (const float*)d_in[20];
  const float* W2  = (const float*)d_in[21];
  const float* al2 = (const float*)d_in[22];
  const float* ar2 = (const float*)d_in[23];
  const float* b2  = (const float*)d_in[24];
  const float* Wres2 = (const float*)d_in[25];
  const float* g2  = (const float*)d_in[26];
  const float* bt2 = (const float*)d_in[27];
  const float* m2  = (const float*)d_in[28];
  const float* v2  = (const float*)d_in[29];
  const float* We  = (const float*)d_in[30];
  const float* be  = (const float*)d_in[31];
  const float* Wc1 = (const float*)d_in[32];
  const float* bc1 = (const float*)d_in[33];
  const float* Wc2 = (const float*)d_in[34];
  const float* bc2 = (const float*)d_in[35];
  const float* Wc3 = (const float*)d_in[36];
  const float* bc3 = (const float*)d_in[37];

  const int N = in_sizes[0] / 128;   // 50000
  const int E = in_sizes[2];         // 500000
  const int Mt = N / 16;             // 3125 row tiles

  // ---- workspace layout: fp32 region, then bf16 region ----
  float* ws = (float*)d_ws;
  float*    feat = ws;                         // N*128
  float*    res  = feat + (size_t)N * 128;     // N*128
  float*    hA   = res  + (size_t)N * 128;     // N*128
  float*    hB   = hA   + (size_t)N * 128;     // N*128
  float*    rst  = hB   + (size_t)N * 128;     // N*128
  float*    el   = rst  + (size_t)N * 128;     // N*8
  float*    er   = el   + (size_t)N * 8;       // N*8
  unsigned* menc = (unsigned*)(er + (size_t)N * 8); // N*8
  float*    sbuf = (float*)(menc + (size_t)N * 8);  // N*8
  float*    ebuf = sbuf + (size_t)N * 8;       // E*8
  float*    exbuf= ebuf + (size_t)E * 8;       // E*8
  __bf16*   abf  = (__bf16*)(exbuf + (size_t)E * 8); // N*128 bf16 (activations)
  __bf16*   ebf  = abf  + (size_t)N * 128;     // E*64 bf16 (edge feats)
  __bf16*   W0p  = ebf  + (size_t)E * 64;      // packed weights (bf16)
  __bf16*   W1p    = W0p   + 128 * 128;
  __bf16*   Wres1p = W1p   + 128 * 64;
  __bf16*   W2p    = Wres1p + 128 * 64;
  __bf16*   Wres2p = W2p   + 64 * 32;
  __bf16*   Wep    = Wres2p + 64 * 32;
  __bf16*   Wc1p   = Wep   + 64 * 128;
  __bf16*   Wc2p   = Wc1p  + 192 * 32;
  float*    out  = (float*)d_out;              // E*2

  const int TB = 256;

  // ---- one-time conversions / weight packing ----
  k_f32_to_bf16<<<cdiv(E * 64, TB), TB, 0, stream>>>(edge_feats, ebf, E * 64);
  k_pack_b<<<cdiv(128 * 128 / 16, TB), TB, 0, stream>>>(W0,    W0p,    128, 128);
  k_pack_b<<<cdiv(128 *  64 / 16, TB), TB, 0, stream>>>(W1,    W1p,    128,  64);
  k_pack_b<<<cdiv(128 *  64 / 16, TB), TB, 0, stream>>>(Wres1, Wres1p, 128,  64);
  k_pack_b<<<cdiv( 64 *  32 / 16, TB), TB, 0, stream>>>(W2,    W2p,     64,  32);
  k_pack_b<<<cdiv( 64 *  32 / 16, TB), TB, 0, stream>>>(Wres2, Wres2p,  64,  32);
  k_pack_b<<<cdiv( 64 * 128 / 16, TB), TB, 0, stream>>>(We,    Wep,     64, 128);
  k_pack_b<<<cdiv(192 *  32 / 16, TB), TB, 0, stream>>>(Wc1,   Wc1p,   192,  32);
  k_pack_b<<<cdiv( 32 *  16 / 16, TB), TB, 0, stream>>>(Wc2,   Wc2p,    32,  16);

  // =================== GAT layer 0: 128 -> 8x16 ===================
  { const int H = 8, D = 16, C = 128;
    k_f32_to_bf16<<<cdiv(N * 128, TB), TB, 0, stream>>>(node_feats, abf, N * 128);
    k_gemm_bf16<<<dim3(cdiv(Mt, 8), C / 16), 256, 0, stream>>>(abf, W0p, nullptr, feat, Mt, 128, C, 0);
    k_attn<<<cdiv(N * H, TB), TB, 0, stream>>>(feat, al0, ar0, el, er, N, H, D);
    k_zero<<<cdiv(N * H, TB), TB, 0, stream>>>((float*)menc, N * H);
    k_zero<<<cdiv(N * H, TB), TB, 0, stream>>>(sbuf, N * H);
    k_zero<<<cdiv(N * C, TB), TB, 0, stream>>>(rst, N * C);
    k_edge_max<<<cdiv(E * H, TB), TB, 0, stream>>>(el, er, src, dst, ebuf, menc, E, H);
    k_edge_expsum<<<cdiv(E * H, TB), TB, 0, stream>>>(ebuf, menc, dst, exbuf, sbuf, E, H);
    k_edge_aggr<<<cdiv(E * H, TB), TB, 0, stream>>>(exbuf, sbuf, src, dst, feat, rst, E, H, D);
    k_finalize<<<cdiv(N * C, TB), TB, 0, stream>>>(rst, node_feats, b0, g0, bt0, m0, v0, hA, N, C, 1);
  }
  // =================== GAT layer 1: 128 -> 8x8 (res proj) ===================
  { const int H = 8, D = 8, C = 64;
    k_f32_to_bf16<<<cdiv(N * 128, TB), TB, 0, stream>>>(hA, abf, N * 128);
    k_gemm_bf16<<<dim3(cdiv(Mt, 8), C / 16), 256, 0, stream>>>(abf, W1p,    nullptr, feat, Mt, 128, C, 0);
    k_gemm_bf16<<<dim3(cdiv(Mt, 8), C / 16), 256, 0, stream>>>(abf, Wres1p, nullptr, res,  Mt, 128, C, 0);
    k_attn<<<cdiv(N * H, TB), TB, 0, stream>>>(feat, al1, ar1, el, er, N, H, D);
    k_zero<<<cdiv(N * H, TB), TB, 0, stream>>>((float*)menc, N * H);
    k_zero<<<cdiv(N * H, TB), TB, 0, stream>>>(sbuf, N * H);
    k_zero<<<cdiv(N * C, TB), TB, 0, stream>>>(rst, N * C);
    k_edge_max<<<cdiv(E * H, TB), TB, 0, stream>>>(el, er, src, dst, ebuf, menc, E, H);
    k_edge_expsum<<<cdiv(E * H, TB), TB, 0, stream>>>(ebuf, menc, dst, exbuf, sbuf, E, H);
    k_edge_aggr<<<cdiv(E * H, TB), TB, 0, stream>>>(exbuf, sbuf, src, dst, feat, rst, E, H, D);
    k_finalize<<<cdiv(N * C, TB), TB, 0, stream>>>(rst, res, b1, g1, bt1, m1, v1, hB, N, C, 1);
  }
  // =================== GAT layer 2: 64 -> 1x32 (res proj, no act) ===================
  { const int H = 1, D = 32, C = 32;
    k_f32_to_bf16<<<cdiv(N * 64, TB), TB, 0, stream>>>(hB, abf, N * 64);
    k_gemm_bf16<<<dim3(cdiv(Mt, 8), C / 16), 256, 0, stream>>>(abf, W2p,    nullptr, feat, Mt, 64, C, 0);
    k_gemm_bf16<<<dim3(cdiv(Mt, 8), C / 16), 256, 0, stream>>>(abf, Wres2p, nullptr, res,  Mt, 64, C, 0);
    k_attn<<<cdiv(N * H, TB), TB, 0, stream>>>(feat, al2, ar2, el, er, N, H, D);
    k_zero<<<cdiv(N * H, TB), TB, 0, stream>>>((float*)menc, N * H);
    k_zero<<<cdiv(N * H, TB), TB, 0, stream>>>(sbuf, N * H);
    k_zero<<<cdiv(N * C, TB), TB, 0, stream>>>(rst, N * C);
    k_edge_max<<<cdiv(E * H, TB), TB, 0, stream>>>(el, er, src, dst, ebuf, menc, E, H);
    k_edge_expsum<<<cdiv(E * H, TB), TB, 0, stream>>>(ebuf, menc, dst, exbuf, sbuf, E, H);
    k_edge_aggr<<<cdiv(E * H, TB), TB, 0, stream>>>(exbuf, sbuf, src, dst, feat, rst, E, H, D);
    k_finalize<<<cdiv(N * C, TB), TB, 0, stream>>>(rst, res, b2, g2, bt2, m2, v2, hA, N, C, 0);
  }
  // =================== fused edge classifier ===================
  k_classifier<<<E / 16, 32, 0, stream>>>(hA, ebf, src, dst, Wep, be,
                                          Wc1p, bc1, Wc2p, bc2, Wc3, bc3, out, E);
}
